// AttentionHead_29781303230587
// MI455X (gfx1250) — matrix operand order repaired and checked by reading, not compiled, not run on hardware
//
#include <hip/hip_runtime.h>
#include <stdint.h>

#define DIM_IN 512
#define DM     64
#define NB     4
#define SEQ    4096
#define BQ     64      // query rows per block (16 per wave, 4 waves)
#define BK     64      // key rows per LDS tile
#define LPAD   8       // LDS row padding (halves) to break bank conflicts

typedef __attribute__((ext_vector_type(16))) _Float16 v16h;
typedef __attribute__((ext_vector_type(8)))  _Float16 v8h;
typedef __attribute__((ext_vector_type(8)))  float    v8f;
typedef __attribute__((ext_vector_type(4)))  float    v4f;
typedef __attribute__((ext_vector_type(4)))  int      v4i;

#if __has_builtin(__builtin_amdgcn_global_load_async_to_lds_b128) && \
    __has_builtin(__builtin_amdgcn_s_wait_asynccnt)
#define USE_ASYNC_LDS 1
#else
#define USE_ASYNC_LDS 0
#endif

#if USE_ASYNC_LDS
typedef v4i __attribute__((address_space(1))) g_v4i;   // global (device) ptr
typedef v4i __attribute__((address_space(3))) l_v4i;   // LDS (shared) ptr
__device__ __forceinline__ void async_b128(const void* g, void* l) {
  // (global src, lds dst, imm offset, imm cpol)
  __builtin_amdgcn_global_load_async_to_lds_b128((g_v4i*)g, (l_v4i*)l, 0, 0);
}
#endif

__device__ __forceinline__ v8f wmma_f16(v16h a, v16h b, v8f c) {
  // 8 args: (neg_a, A, neg_b, B, c_mod, C, reuse_a, reuse_b)
  return __builtin_amdgcn_wmma_f32_16x16x32_f16(false, a, false, b, (short)0, c,
                                                false, false);
}

// ---------------------------------------------------------------------------
// Kernel 1: fused Q/K/V projection GEMMs  [16384,512] x [512,64] + bias -> f16
// One wave computes a 16x16 output tile; 4 waves/block cover all 64 columns of
// one 16-row tile. Q,V stored row-major [b*S, 64]; K stored TRANSPOSED
// [b][d][key] so the attention kernel's K^T LDS tile is a contiguous copy.
// ---------------------------------------------------------------------------
__global__ __launch_bounds__(128)
void proj_qkv_kernel(const float* __restrict__ q, const float* __restrict__ k,
                     const float* __restrict__ v,
                     const float* __restrict__ Wq, const float* __restrict__ bq,
                     const float* __restrict__ Wk, const float* __restrict__ bk,
                     const float* __restrict__ Wv, const float* __restrict__ bv,
                     _Float16* __restrict__ outq, _Float16* __restrict__ outkT,
                     _Float16* __restrict__ outv) {
  const int ROWTILES = (NB * SEQ) / 16;              // 1024
  const int rowTile  = blockIdx.x % ROWTILES;
  const int mi       = blockIdx.x / ROWTILES;        // 0:q 1:k 2:v

  const float* X; const float* W; const float* bias;
  if (mi == 0)      { X = q; W = Wq; bias = bq; }
  else if (mi == 1) { X = k; W = Wk; bias = bk; }
  else              { X = v; W = Wv; bias = bv; }

  const int lane = threadIdx.x & 31;
  const int wv   = threadIdx.x >> 5;
  const int hi   = lane >> 4;          // lane half
  const int l15  = lane & 15;
  const int n0   = wv * 16;            // output column tile
  const int m    = rowTile * 16 + l15; // A-matrix row for this lane

  v8f c = {};
#pragma unroll 4
  for (int kb = 0; kb < DIM_IN; kb += 32) {
    // A fragment (16x32 f16): lane<16 holds K={0..7,16..23}, lane>=16 {8..15,24..31}
    v16h a;
    {
      const float* p0 = X + (size_t)m * DIM_IN + kb + hi * 8;
      v4f f0 = *(const v4f*)(p0);
      v4f f1 = *(const v4f*)(p0 + 4);
      v4f f2 = *(const v4f*)(p0 + 16);
      v4f f3 = *(const v4f*)(p0 + 20);
#pragma unroll
      for (int j = 0; j < 4; ++j) {
        a[j]      = (_Float16)f0[j];
        a[4 + j]  = (_Float16)f1[j];
        a[8 + j]  = (_Float16)f2[j];
        a[12 + j] = (_Float16)f3[j];
      }
    }
    // B fragment (32x16 f16): lane holds column n0+l15, K={0..7,16..23} (lo) / {8..15,24..31} (hi)
    v16h bf;
    {
      const float* p = W + (size_t)(kb + hi * 8) * DM + n0 + l15;
#pragma unroll
      for (int j = 0; j < 8; ++j) {
        bf[j]     = (_Float16)p[(size_t)j * DM];
        bf[8 + j] = (_Float16)p[(size_t)(16 + j) * DM];
      }
    }
    c = wmma_f16(a, bf, c);
  }

  const float bb = bias[n0 + l15];     // same column for all 8 rows this lane owns
  if (mi == 1) {
    // Transposed store: this lane's 8 C values are column (n0+l15) of the tile
    // at contiguous rows (keys) rowTile*16 + 8*hi + [0..8)  -> one v8h store.
    v8h pk;
#pragma unroll
    for (int r = 0; r < 8; ++r) pk[r] = (_Float16)(c[r] + bb);
    const int flat  = rowTile * 16;
    const int batch = flat / SEQ;
    const int key   = (flat % SEQ) + 8 * hi;
    *(v8h*)&outkT[((size_t)batch * DM + n0 + l15) * SEQ + key] = pk;
  } else {
    _Float16* out = (mi == 0) ? outq : outv;
#pragma unroll
    for (int r = 0; r < 8; ++r) {
      const int M = rowTile * 16 + r + 8 * hi;   // C layout: M = r + 8*half
      out[(size_t)M * DM + n0 + l15] = (_Float16)(c[r] + bb);
    }
  }
}

// ---------------------------------------------------------------------------
// Kernel 2: flash attention. Block = 128 threads (4 waves); each wave owns 16
// query rows; K^T tile + V tile staged in LDS (async to LDS when available)
// and shared by the 4 waves. 16 wmma_f32_16x16x32_f16 per wave per key tile.
// ---------------------------------------------------------------------------
__global__ __launch_bounds__(128)
void flash_attn_kernel(const _Float16* __restrict__ qh,
                       const _Float16* __restrict__ khT,  // [b][d][key]
                       const _Float16* __restrict__ vh,   // [b][key][d]
                       float* __restrict__ out) {
  __shared__ _Float16 KTs[DM][BK + LPAD];     // K^T: [d][key]
  __shared__ _Float16 Vs[BK][DM + LPAD];      // V  : [key][d]
  __shared__ _Float16 Ps[4][16][BK + LPAD];   // per-wave P scratch (C->A relayout)

  const int b    = blockIdx.y;
  const int q0   = blockIdx.x * BQ;
  const int tid  = threadIdx.x;
  const int lane = tid & 31;
  const int wv   = tid >> 5;
  const int hi   = lane >> 4;
  const int l15  = lane & 15;

  // Load persistent Q fragments (16 rows x 64 dims -> two 16x32 A fragments)
  const _Float16* qbase = qh + (size_t)(b * SEQ + q0 + wv * 16) * DM;
  v16h aq[2];
#pragma unroll
  for (int c = 0; c < 2; ++c) {
    const _Float16* p = qbase + (size_t)l15 * DM + c * 32 + hi * 8;
    v8h h0 = *(const v8h*)(p);
    v8h h1 = *(const v8h*)(p + 16);
#pragma unroll
    for (int j = 0; j < 8; ++j) { aq[c][j] = h0[j]; aq[c][8 + j] = h1[j]; }
  }

  float mrow[8], lrow[8];
  v8f acc[4] = {};
#pragma unroll
  for (int r = 0; r < 8; ++r) { mrow[r] = -3.0e38f; lrow[r] = 0.f; }

  const _Float16* ktb = khT + (size_t)b * DM * SEQ;   // row d stride = SEQ
  const _Float16* vb  = vh  + (size_t)b * SEQ * DM;

  // per-thread tile-copy coordinates (2 threads per row, 32 halves each)
  const int row = tid >> 1;            // K^T: d-row   / V: key-row
  const int c0  = (tid & 1) * 32;      // K^T: key off / V: d off

  for (int kt = 0; kt < SEQ; kt += BK) {
    // ---- cooperative tile fill: K^T and V rows are contiguous copies ----
    {
      const _Float16* srck = ktb + (size_t)row * SEQ + kt + c0;
      const _Float16* srcv = vb + (size_t)(kt + row) * DM + c0;
#if USE_ASYNC_LDS
#pragma unroll
      for (int j = 0; j < 4; ++j) {
        async_b128(srck + j * 8, &KTs[row][c0 + j * 8]);
        async_b128(srcv + j * 8, &Vs[row][c0 + j * 8]);
      }
      __builtin_amdgcn_s_wait_asynccnt(0);
#else
#pragma unroll
      for (int j = 0; j < 4; ++j) {
        *(v8h*)&KTs[row][c0 + j * 8] = *(const v8h*)(srck + j * 8);
        *(v8h*)&Vs[row][c0 + j * 8]  = *(const v8h*)(srcv + j * 8);
      }
#endif
      if (kt + BK < SEQ) {                 // global_prefetch_b8 next tile
        __builtin_prefetch(srck + BK, 0, 1);
        __builtin_prefetch(srcv + (size_t)BK * DM, 0, 1);
      }
    }
    __syncthreads();

    // ---- scores: S = Q (16x64) * K^T (64x64) -> four 16x16 f32 tiles ----
    float sv[4][8];
#pragma unroll
    for (int nf = 0; nf < 4; ++nf) {
      v8f s = {};
#pragma unroll
      for (int c = 0; c < 2; ++c) {
        const _Float16* p = &KTs[c * 32 + hi * 8][nf * 16 + l15];
        v16h bf;
#pragma unroll
        for (int j = 0; j < 8; ++j) { bf[j] = p[j * (BK + LPAD)]; bf[8 + j] = p[(16 + j) * (BK + LPAD)]; }
        s = wmma_f16(aq[c], bf, s);
      }
#pragma unroll
      for (int r = 0; r < 8; ++r) sv[nf][r] = s[r] * 0.125f;  // 1/sqrt(64)
    }

    // ---- online softmax over this 16x64 block ----
#pragma unroll
    for (int r = 0; r < 8; ++r) {
      float mx = fmaxf(fmaxf(sv[0][r], sv[1][r]), fmaxf(sv[2][r], sv[3][r]));
#pragma unroll
      for (int off = 1; off < 16; off <<= 1)          // reduce across lane half
        mx = fmaxf(mx, __shfl_xor(mx, off, 32));
      const float mnew  = fmaxf(mrow[r], mx);
      const float alpha = __expf(mrow[r] - mnew);
      mrow[r] = mnew;
      float rs = 0.f;
#pragma unroll
      for (int nf = 0; nf < 4; ++nf) {
        const float p = __expf(sv[nf][r] - mnew);
        sv[nf][r] = p;
        rs += p;
      }
#pragma unroll
      for (int off = 1; off < 16; off <<= 1)
        rs += __shfl_xor(rs, off, 32);
      lrow[r] = lrow[r] * alpha + rs;
#pragma unroll
      for (int d = 0; d < 4; ++d) acc[d][r] *= alpha;
    }

    // ---- C-layout -> A-layout for P via per-wave LDS scratch ----
#pragma unroll
    for (int nf = 0; nf < 4; ++nf)
#pragma unroll
      for (int r = 0; r < 8; ++r)
        Ps[wv][r + 8 * hi][nf * 16 + l15] = (_Float16)sv[nf][r];
    v16h ap[2];
#pragma unroll
    for (int c = 0; c < 2; ++c) {          // wave-local RAW; dscnt-waited
      const _Float16* p = &Ps[wv][l15][c * 32 + hi * 8];
      v8h h0 = *(const v8h*)p;
      v8h h1 = *(const v8h*)(p + 16);
#pragma unroll
      for (int j = 0; j < 8; ++j) { ap[c][j] = h0[j]; ap[c][8 + j] = h1[j]; }
    }

    // ---- acc += P (16x64) * V (64x64) ----
#pragma unroll
    for (int df = 0; df < 4; ++df) {
#pragma unroll
      for (int c = 0; c < 2; ++c) {
        const _Float16* p = &Vs[c * 32 + hi * 8][df * 16 + l15];
        v16h bf;
#pragma unroll
        for (int j = 0; j < 8; ++j) { bf[j] = p[j * (DM + LPAD)]; bf[8 + j] = p[(16 + j) * (DM + LPAD)]; }
        acc[df] = wmma_f16(ap[c], bf, acc[df]);
      }
    }
    __syncthreads();
  }

  // ---- epilogue: out = acc / l ----
  float* obase = out + (size_t)(b * SEQ + q0 + wv * 16) * DM;
#pragma unroll
  for (int df = 0; df < 4; ++df)
#pragma unroll
    for (int r = 0; r < 8; ++r) {
      const int M = r + 8 * hi;
      obase[(size_t)M * DM + df * 16 + l15] = acc[df][r] / lrow[r];
    }
}

extern "C" void kernel_launch(void* const* d_in, const int* in_sizes, int n_in,
                              void* d_out, int out_size, void* d_ws, size_t ws_size,
                              hipStream_t stream) {
  (void)in_sizes; (void)n_in; (void)out_size; (void)ws_size;
  const float* q  = (const float*)d_in[0];
  const float* k  = (const float*)d_in[1];
  const float* v  = (const float*)d_in[2];
  const float* Wq = (const float*)d_in[3];
  const float* bq = (const float*)d_in[4];
  const float* Wk = (const float*)d_in[5];
  const float* bk = (const float*)d_in[6];
  const float* Wv = (const float*)d_in[7];
  const float* bv = (const float*)d_in[8];

  const size_t elems = (size_t)NB * SEQ * DM;        // 1M halves per tensor
  _Float16* qh  = (_Float16*)d_ws;
  _Float16* khT = qh + elems;                        // [b][d][key]
  _Float16* vh  = khT + elems;

  proj_qkv_kernel<<<dim3(3 * (NB * SEQ / 16)), 128, 0, stream>>>(
      q, k, v, Wq, bq, Wk, bk, Wv, bv, qh, khT, vh);
  flash_attn_kernel<<<dim3(SEQ / BQ, NB), 128, 0, stream>>>(
      qh, khT, vh, (float*)d_out);
}
